// LSTMCell_730144441065
// MI455X (gfx1250) — matrix operand order repaired and checked by reading, not compiled, not run on hardware
//
#include <hip/hip_runtime.h>
#include <stdint.h>

// ---------------------------------------------------------------------------
// LSTM cell, MI455X (gfx1250): bf16 WMMA GEMM (f32 accum) + fused activations
//   gates = x @ w_ih^T + hx @ w_hh^T + (b_ih + b_hh)
//   i,f,o = sigmoid; g = tanh; cy = f*cx + i*g; hy = o*tanh(cy)
// d_out = [hy (4096*2048) | cy (4096*2048)] fp32
//
// Round 3: all B fragments preloaded into independent registers so the
// ds_load clause can run ahead of the WMMA burst with partial DScnt waits
// (round 2 recycled one 8-VGPR slot and serialized on s_wait_dscnt 0x0).
// ---------------------------------------------------------------------------

typedef __attribute__((ext_vector_type(16))) __bf16 v16bf;
typedef __attribute__((ext_vector_type(8)))  __bf16 v8bf;
typedef __attribute__((ext_vector_type(8)))  float  v8f;

#define B_ROWS 4096
#define IN_DIM 2048
#define H_DIM  2048

#define BM 128   // rows per workgroup
#define BN 64    // h-columns per workgroup (x4 gates internally)
#define BK 32    // K step == WMMA K
#define LDS_STRIDE 40   // 32 elements + 8 pad (80B rows, 16B aligned)
#define NSTEPS 128      // 2 passes x (2048/32)

__device__ __forceinline__ float fast_sigmoid(float x) {
    return 1.0f / (1.0f + __expf(-x));
}
__device__ __forceinline__ float fast_tanh(float x) {
    // branch-free; saturates correctly via exp under/overflow
    return 2.0f / (1.0f + __expf(-2.0f * x)) - 1.0f;
}

__device__ __forceinline__ unsigned short f32_to_bf16_rne(float f) {
    unsigned int u = __builtin_bit_cast(unsigned int, f);
    unsigned int r = u + 0x7FFFu + ((u >> 16) & 1u);
    return (unsigned short)(r >> 16);
}

// async global -> LDS, 16B per lane, saddr addressing (ASYNCcnt tracked)
__device__ __forceinline__ void async_ld128(uint32_t lds_addr, uint32_t voff,
                                            const unsigned short* base) {
    asm volatile("global_load_async_to_lds_b128 %0, %1, %2"
                 :: "v"(lds_addr), "v"(voff), "s"(base) : "memory");
}
__device__ __forceinline__ void wait_async0() {
    asm volatile("s_wait_asynccnt 0" ::: "memory");
}

// --------------------------- fp32 -> bf16 pass -----------------------------
__global__ __launch_bounds__(256)
void convert_f32_bf16(const float* __restrict__ src,
                      unsigned short* __restrict__ dst, int n) {
    int i = (blockIdx.x * 256 + threadIdx.x) * 8;
    if (i + 7 < n) {
        float4 a = *(const float4*)(src + i);
        float4 b = *(const float4*)(src + i + 4);
        unsigned short t[8];
        t[0] = f32_to_bf16_rne(a.x); t[1] = f32_to_bf16_rne(a.y);
        t[2] = f32_to_bf16_rne(a.z); t[3] = f32_to_bf16_rne(a.w);
        t[4] = f32_to_bf16_rne(b.x); t[5] = f32_to_bf16_rne(b.y);
        t[6] = f32_to_bf16_rne(b.z); t[7] = f32_to_bf16_rne(b.w);
        *(uint4*)(dst + i) = *(uint4*)t;
    }
}

// ------------------- fused dual-GEMM + LSTM activations --------------------
__global__ __launch_bounds__(256)
void lstm_gemm_fused(const unsigned short* __restrict__ xb,
                     const unsigned short* __restrict__ hb,
                     const unsigned short* __restrict__ wihb,
                     const unsigned short* __restrict__ whhb,
                     const float* __restrict__ cx,
                     const float* __restrict__ b_ih,
                     const float* __restrict__ b_hh,
                     float* __restrict__ out) {
    // double-buffered LDS tiles
    __shared__ __align__(16) unsigned short ldsA[2][BM * LDS_STRIDE];       // 2 x 10 KB
    __shared__ __align__(16) unsigned short ldsB[2][4 * BN * LDS_STRIDE];   // 2 x 20 KB

    const int tid  = threadIdx.x;
    const int lane = tid & 31;
    const int w    = tid >> 5;     // wave 0..7
    const int wm   = w >> 1;       // 0..3 -> 32-row group
    const int wn   = w & 1;        // 0..1 -> 32-col group
    const int browBase = blockIdx.y * BM;
    const int bcolBase = blockIdx.x * BN;

    // ---- per-thread async staging addresses (byte units) ----
    uint32_t aLds[2], aVoff[2];
    #pragma unroll
    for (int i = 0; i < 2; ++i) {               // A: 512 x 16B chunks
        int c = tid + i * 256;
        int row = c >> 2, q = c & 3;
        aLds[i]  = (uint32_t)(row * LDS_STRIDE + q * 8) * 2u;
        aVoff[i] = (uint32_t)((browBase + row) * IN_DIM + q * 8) * 2u;
    }
    uint32_t bLds[4], bVoff[4];
    #pragma unroll
    for (int i = 0; i < 4; ++i) {               // B: 1024 x 16B chunks
        int c = tid + i * 256;
        int row = c >> 2, q = c & 3;
        int gate = row >> 6, col = row & 63;
        bLds[i]  = (uint32_t)(row * LDS_STRIDE + q * 8) * 2u;
        bVoff[i] = (uint32_t)((gate * H_DIM + bcolBase + col) * IN_DIM + q * 8) * 2u;
    }

    auto stage = [&](int s) {
        int pass = s >> 6;                       // 0: (x, w_ih)  1: (hx, w_hh)
        const unsigned short* aS = pass ? hb : xb;
        const unsigned short* wS = pass ? whhb : wihb;
        uint32_t kb = (uint32_t)(s & 63) * (BK * 2u);   // K byte offset
        int b = s & 1;
        uint32_t aBase = (uint32_t)(uintptr_t)(&ldsA[b][0]);
        uint32_t bBase = (uint32_t)(uintptr_t)(&ldsB[b][0]);
        #pragma unroll
        for (int i = 0; i < 2; ++i) async_ld128(aBase + aLds[i], aVoff[i] + kb, aS);
        #pragma unroll
        for (int i = 0; i < 4; ++i) async_ld128(bBase + bLds[i], bVoff[i] + kb, wS);
    };

    v8f acc[2][2][4];              // [mtile][ntile][gate]
    #pragma unroll
    for (int mt = 0; mt < 2; ++mt)
        #pragma unroll
        for (int nt = 0; nt < 2; ++nt)
            #pragma unroll
            for (int g = 0; g < 4; ++g)
                acc[mt][nt][g] = (v8f){0.f,0.f,0.f,0.f,0.f,0.f,0.f,0.f};

    // prologue: fill buffer 0
    stage(0);
    wait_async0();
    __syncthreads();

    #pragma unroll 1
    for (int s = 0; s < NSTEPS; ++s) {
        if (s + 1 < NSTEPS) stage(s + 1);        // overlap copy with compute
        const int cur = s & 1;

        // ---- issue ALL fragment loads first (independent registers) ----
        // A fragments (16x32 bf16): lanes0-15 K0..7/K16..23, lanes16-31 +8
        v16bf aFrag[2];
        #pragma unroll
        for (int mt = 0; mt < 2; ++mt) {
            int row  = wm * 32 + mt * 16 + (lane & 15);
            int koff = (lane >> 4) * 8;                 // elements
            v8bf lo = *(const v8bf*)(&ldsA[cur][row * LDS_STRIDE + koff]);
            v8bf hi = *(const v8bf*)(&ldsA[cur][row * LDS_STRIDE + koff + 16]);
            aFrag[mt] = __builtin_shufflevector(lo, hi,
                0,1,2,3,4,5,6,7,8,9,10,11,12,13,14,15);
        }
        // B fragments (32x16 bf16): lanes0-15 K0..7/K16..23, lanes16-31 +8
        v16bf bFrag[8];                                  // [g*2+nt]
        #pragma unroll
        for (int g = 0; g < 4; ++g) {
            #pragma unroll
            for (int nt = 0; nt < 2; ++nt) {
                int row  = g * 64 + wn * 32 + nt * 16 + (lane & 15);
                int koff = (lane >> 4) * 8;             // elements
                v8bf lo = *(const v8bf*)(&ldsB[cur][row * LDS_STRIDE + koff]);
                v8bf hi = *(const v8bf*)(&ldsB[cur][row * LDS_STRIDE + koff + 16]);
                bFrag[g * 2 + nt] = __builtin_shufflevector(lo, hi,
                    0,1,2,3,4,5,6,7,8,9,10,11,12,13,14,15);
            }
        }

        // ---- WMMA burst: 16 matrix ops consuming the preloaded fragments ----
        #pragma unroll
        for (int g = 0; g < 4; ++g)
            #pragma unroll
            for (int nt = 0; nt < 2; ++nt)
                #pragma unroll
                for (int mt = 0; mt < 2; ++mt)
                    acc[mt][nt][g] = __builtin_amdgcn_wmma_f32_16x16x32_bf16(
                        false, aFrag[mt], false, bFrag[g * 2 + nt],
                        (short)0, acc[mt][nt][g], false, false);

        wait_async0();        // next buffer resident (hidden behind compute)
        __syncthreads();      // all waves done reading cur buffer
    }

    // ---- fused epilogue: all 4 gates for each (row,col) live in this wave ----
    // C/D layout: VGPR r: lanes0-15 -> M=r, N=lane; lanes16-31 -> M=8+r, N=lane-16
    #pragma unroll
    for (int nt = 0; nt < 2; ++nt) {
        int gcol = bcolBase + wn * 32 + nt * 16 + (lane & 15);
        float bsum[4];
        #pragma unroll
        for (int g = 0; g < 4; ++g)
            bsum[g] = b_ih[g * H_DIM + gcol] + b_hh[g * H_DIM + gcol];
        #pragma unroll
        for (int mt = 0; mt < 2; ++mt) {
            int rbase = browBase + wm * 32 + mt * 16 + ((lane >> 4) << 3);
            #pragma unroll
            for (int e = 0; e < 8; ++e) {
                int grow = rbase + e;
                float gi = fast_sigmoid(acc[mt][nt][0][e] + bsum[0]);
                float gf = fast_sigmoid(acc[mt][nt][1][e] + bsum[1]);
                float gg = fast_tanh  (acc[mt][nt][2][e] + bsum[2]);
                float go = fast_sigmoid(acc[mt][nt][3][e] + bsum[3]);
                float c0 = cx[(size_t)grow * H_DIM + gcol];
                float cyv = gf * c0 + gi * gg;
                float hyv = go * fast_tanh(cyv);
                out[(size_t)grow * H_DIM + gcol] = hyv;
                out[(size_t)B_ROWS * H_DIM + (size_t)grow * H_DIM + gcol] = cyv;
            }
        }
    }
}

// ---------------------------------------------------------------------------
extern "C" void kernel_launch(void* const* d_in, const int* in_sizes, int n_in,
                              void* d_out, int out_size, void* d_ws, size_t ws_size,
                              hipStream_t stream) {
    (void)in_sizes; (void)n_in; (void)out_size; (void)ws_size;
    const float* x   = (const float*)d_in[0];
    const float* hx  = (const float*)d_in[1];
    const float* cx  = (const float*)d_in[2];
    const float* wih = (const float*)d_in[3];
    const float* whh = (const float*)d_in[4];
    const float* bih = (const float*)d_in[5];
    const float* bhh = (const float*)d_in[6];
    float* out = (float*)d_out;

    // bf16 staging in workspace: x(16MB) | hx(16MB) | w_ih(32MB) | w_hh(32MB)
    unsigned short* xb   = (unsigned short*)d_ws;
    unsigned short* hb   = xb  + (size_t)B_ROWS * IN_DIM;
    unsigned short* wihb = hb  + (size_t)B_ROWS * H_DIM;
    unsigned short* whhb = wihb + (size_t)4 * H_DIM * IN_DIM;

    const int nx = B_ROWS * IN_DIM;      // 8,388,608
    const int nw = 4 * H_DIM * IN_DIM;   // 16,777,216
    dim3 blk(256);
    convert_f32_bf16<<<dim3(nx / 2048), blk, 0, stream>>>(x,   xb,   nx);
    convert_f32_bf16<<<dim3(nx / 2048), blk, 0, stream>>>(hx,  hb,   nx);
    convert_f32_bf16<<<dim3(nw / 2048), blk, 0, stream>>>(wih, wihb, nw);
    convert_f32_bf16<<<dim3(nw / 2048), blk, 0, stream>>>(whh, whhb, nw);

    lstm_gemm_fused<<<dim3(H_DIM / BN, B_ROWS / BM), blk, 0, stream>>>(
        xb, hb, wihb, whhb, cx, bih, bhh, out);
}
